// Net_51608327029501
// MI455X (gfx1250) — matrix operand order repaired and checked by reading, not compiled, not run on hardware
//
#include <hip/hip_runtime.h>

// ---------------------------------------------------------------------------
// GCN link-predictor for MI455X (gfx1250, wave32).
//   - dense matmuls on v_wmma_f32_16x16x32_bf16 (fp32 -> bf16 in-register)
//   - edge aggregation via float4 gathers + coalesced global_atomic_add_f32
//     (bandwidth bound: ~3.5 GB gather/scatter dominates; ~12 GF dense is noise)
//   - decode MLP fully fused per 16-edge tile (WMMA + shfl reduction)
// ---------------------------------------------------------------------------

typedef __attribute__((ext_vector_type(16))) __bf16 v16bf;
typedef __attribute__((ext_vector_type(8)))  float  v8f;

#define WMMA_BF16(a, b, c) \
  __builtin_amdgcn_wmma_f32_16x16x32_bf16(false, (a), false, (b), (short)0, (c), false, false)

__device__ __forceinline__ v16bf make_frag(float4 a0, float4 a1, float4 b0, float4 b1) {
  v16bf v;
  v[0]  = (__bf16)a0.x; v[1]  = (__bf16)a0.y; v[2]  = (__bf16)a0.z; v[3]  = (__bf16)a0.w;
  v[4]  = (__bf16)a1.x; v[5]  = (__bf16)a1.y; v[6]  = (__bf16)a1.z; v[7]  = (__bf16)a1.w;
  v[8]  = (__bf16)b0.x; v[9]  = (__bf16)b0.y; v[10] = (__bf16)b0.z; v[11] = (__bf16)b0.w;
  v[12] = (__bf16)b1.x; v[13] = (__bf16)b1.y; v[14] = (__bf16)b1.z; v[15] = (__bf16)b1.w;
  return v;
}

// A fragment (16x32 bf16, row-major source): lane holds row M = lane&15.
// Element j -> K = kb+koff+j (j<8), K = kb+16+koff+(j-8) (j>=8), koff = (lane>>4)*8.
__device__ __forceinline__ v16bf load_a_frag(const float* __restrict__ rowp, int kb, int koff) {
  const float* p = rowp + kb + koff;
  float4 a0 = *(const float4*)(p);
  float4 a1 = *(const float4*)(p + 4);
  float4 b0 = *(const float4*)(p + 16);
  float4 b1 = *(const float4*)(p + 20);
  return make_frag(a0, a1, b0, b1);
}

// B fragment (32x16 bf16, row-major [K x F] source): lane holds col N = lane&15.
// Element j -> K = kb+khalf/2+j (j<8), K = kb+16+khalf/2+(j-8) (j>=8), khalf = (lane>>4)*16.
__device__ __forceinline__ v16bf load_b_frag(const float* __restrict__ W, int ldw,
                                             int kb, int khalf, int col) {
  v16bf v;
#pragma unroll
  for (int j = 0; j < 16; ++j) v[j] = (__bf16)W[(size_t)(kb + (khalf >> 1) + j + (j >= 8 ? 8 : 0)) * ldw + col];
  return v;
}

// ---------------------------------------------------------------------------
// deg[dst] += 1 over edges (float accumulation; finalize applies +1 and rsqrt)
// ---------------------------------------------------------------------------
__global__ void degree_kernel(const int* __restrict__ dst, float* __restrict__ deg, int ne) {
  int i = blockIdx.x * blockDim.x + threadIdx.x;
  if (i < ne) atomicAdd(&deg[dst[i]], 1.0f);
}

__global__ void finalize_dinv_kernel(float* __restrict__ deg, int n) {
  int i = blockIdx.x * blockDim.x + threadIdx.x;
  if (i < n) deg[i] = rsqrtf(deg[i] + 1.0f);
}

// ---------------------------------------------------------------------------
// out[nrows x F] = A[nrows x K] @ W[K x F]   (bf16 WMMA, f32 accumulate)
// one 16x16 tile per wave; 4 waves / block; grid = (row tiles / 4, F/16)
// ---------------------------------------------------------------------------
template <int K, int F>
__global__ void gemm_wmma_kernel(const float* __restrict__ A, const float* __restrict__ W,
                                 float* __restrict__ out, int nrows) {
  const int lane = threadIdx.x & 31;
  const int wave = threadIdx.x >> 5;
  const int tm   = blockIdx.x * 4 + wave;
  if (tm * 16 >= nrows) return;
  const int tn    = blockIdx.y;
  const int koff  = (lane >> 4) * 8;
  const int khalf = (lane >> 4) * 16;
  const int col   = tn * 16 + (lane & 15);
  int arow = tm * 16 + (lane & 15);
  if (arow >= nrows) arow = nrows - 1;        // clamp; tail rows never stored
  const float* rowp = A + (size_t)arow * K;

  v8f c = {};
#pragma unroll
  for (int kb = 0; kb < K; kb += 32) {
    v16bf a = load_a_frag(rowp, kb, koff);
    v16bf b = load_b_frag(W, F, kb, khalf, col);
    c = WMMA_BF16(a, b, c);
  }

  const int rbase = tm * 16 + (lane >> 4) * 8;
#pragma unroll
  for (int r = 0; r < 8; ++r) {
    int rr = rbase + r;
    if (rr < nrows) out[(size_t)rr * F + col] = c[r];
  }
}

// ---------------------------------------------------------------------------
// agg[dst] += xw[src] * dinv[src]*dinv[dst]
// one thread per (edge, 4-feature group): one b128 gather + 4 f32 atomics;
// index/dinv loads amortized 4x; consecutive threads -> consecutive features.
// ---------------------------------------------------------------------------
template <int F>
__global__ void aggregate_kernel(const float* __restrict__ xw, const float* __restrict__ dinv,
                                 const int* __restrict__ src, const int* __restrict__ dst,
                                 float* __restrict__ agg, int ne) {
  constexpr int G = F / 4;
  long long total  = (long long)ne * G;
  long long stride = (long long)gridDim.x * blockDim.x;
  for (long long i = (long long)blockIdx.x * blockDim.x + threadIdx.x; i < total; i += stride) {
    int e = (int)(i / G);
    int f = (int)(i - (long long)e * G) * 4;
    int s = src[e], d = dst[e];
    float w = dinv[s] * dinv[d];
    float4 v = *(const float4*)(xw + (size_t)s * F + f);
    float* ap = agg + (size_t)d * F + f;
    atomicAdd(ap + 0, v.x * w);
    atomicAdd(ap + 1, v.y * w);
    atomicAdd(ap + 2, v.z * w);
    atomicAdd(ap + 3, v.w * w);
  }
}

// ---------------------------------------------------------------------------
// out = [relu]( agg + xw * dinv^2 + bias )  (float4 elementwise; in-place safe)
// ---------------------------------------------------------------------------
template <int F, bool RELU>
__global__ void combine_kernel(const float* __restrict__ agg, const float* __restrict__ xw,
                               const float* __restrict__ dinv, const float* __restrict__ bias,
                               float* __restrict__ out, int n) {
  constexpr int G = F / 4;
  long long total  = (long long)n * G;
  long long stride = (long long)gridDim.x * blockDim.x;
  for (long long i = (long long)blockIdx.x * blockDim.x + threadIdx.x; i < total; i += stride) {
    int node = (int)(i / G);
    int f    = (int)(i - (long long)node * G) * 4;
    float di  = dinv[node];
    float di2 = di * di;
    float4 a = *(const float4*)(agg + i * 4);
    float4 x = *(const float4*)(xw + i * 4);
    float4 b = *(const float4*)(bias + f);
    float4 r;
    r.x = a.x + x.x * di2 + b.x;
    r.y = a.y + x.y * di2 + b.y;
    r.z = a.z + x.z * di2 + b.z;
    r.w = a.w + x.w * di2 + b.w;
    if (RELU) {
      r.x = fmaxf(r.x, 0.0f); r.y = fmaxf(r.y, 0.0f);
      r.z = fmaxf(r.z, 0.0f); r.w = fmaxf(r.w, 0.0f);
    }
    *(float4*)(out + i * 4) = r;
  }
}

// ---------------------------------------------------------------------------
// decode: per 16-edge tile (one wave):
//   A1 = [z[s] | z[d]] (16x128), A2 = [z[d] | z[s]]
//   h  = relu(A @ W1 + b1) (16x64 via 4 WMMA n-tiles x 4 k-steps)
//   o  = h @ W2 + b2  (per-lane weighted sum + 16-lane shfl_xor reduce)
//   out[0][m] = -0.5*(o1+o2), out[1][m] = +0.5*(o1+o2)
// Fragments are shared between the two concat orders (just K-block swap).
// ---------------------------------------------------------------------------
__global__ void decode_kernel(const float* __restrict__ z, const int* __restrict__ eli,
                              const float* __restrict__ W1, const float* __restrict__ b1,
                              const float* __restrict__ W2, const float* __restrict__ b2,
                              float* __restrict__ out, int nel) {
  const int lane = threadIdx.x & 31;
  const int wave = threadIdx.x >> 5;
  const int tile = blockIdx.x * 4 + wave;
  if (tile * 16 >= nel) return;

  int m = tile * 16 + (lane & 15);
  if (m >= nel) m = nel - 1;
  const int s = eli[m];
  const int d = eli[nel + m];
  const float* zs = z + (size_t)s * 64;
  const float* zd = z + (size_t)d * 64;
  const int koff  = (lane >> 4) * 8;
  const int khalf = (lane >> 4) * 16;

  // A fragments reused by both directions
  v16bf as0  = load_a_frag(zs, 0,  koff);
  v16bf as32 = load_a_frag(zs, 32, koff);
  v16bf ad0  = load_a_frag(zd, 0,  koff);
  v16bf ad32 = load_a_frag(zd, 32, koff);

  float p1[8], p2[8];
#pragma unroll
  for (int r = 0; r < 8; ++r) { p1[r] = 0.0f; p2[r] = 0.0f; }

#pragma unroll
  for (int t = 0; t < 4; ++t) {
    const int col = t * 16 + (lane & 15);
    v16bf b0  = load_b_frag(W1, 64, 0,  khalf, col);
    v16bf b32 = load_b_frag(W1, 64, 32, khalf, col);
    v16bf b64 = load_b_frag(W1, 64, 64, khalf, col);
    v16bf b96 = load_b_frag(W1, 64, 96, khalf, col);

    v8f c1 = {};
    c1 = WMMA_BF16(as0,  b0,  c1);
    c1 = WMMA_BF16(as32, b32, c1);
    c1 = WMMA_BF16(ad0,  b64, c1);
    c1 = WMMA_BF16(ad32, b96, c1);

    v8f c2 = {};
    c2 = WMMA_BF16(ad0,  b0,  c2);
    c2 = WMMA_BF16(ad32, b32, c2);
    c2 = WMMA_BF16(as0,  b64, c2);
    c2 = WMMA_BF16(as32, b96, c2);

    const float bb  = b1[col];
    const float w2v = W2[col];
#pragma unroll
    for (int r = 0; r < 8; ++r) {
      p1[r] += fmaxf(c1[r] + bb, 0.0f) * w2v;
      p2[r] += fmaxf(c2[r] + bb, 0.0f) * w2v;
    }
  }

  // reduce across the 16 lanes of each half (rows live per half: 0-7 / 8-15)
#pragma unroll
  for (int off = 8; off >= 1; off >>= 1) {
#pragma unroll
    for (int r = 0; r < 8; ++r) {
      p1[r] += __shfl_xor(p1[r], off);
      p2[r] += __shfl_xor(p2[r], off);
    }
  }

  if ((lane & 15) == 0) {                 // lane 0 -> rows 0..7, lane 16 -> rows 8..15
    const int base = tile * 16 + (lane >> 4) * 8;
    const float bias2 = b2[0];
#pragma unroll
    for (int r = 0; r < 8; ++r) {
      int mrow = base + r;
      if (mrow < nel) {
        float v = 0.5f * (p1[r] + p2[r]) + bias2;
        out[mrow]               = -v;
        out[(size_t)nel + mrow] =  v;
      }
    }
  }
}

static inline int ceil_div(int a, int b) { return (a + b - 1) / b; }

extern "C" void kernel_launch(void* const* d_in, const int* in_sizes, int n_in,
                              void* d_out, int out_size, void* d_ws, size_t ws_size,
                              hipStream_t stream) {
  const float* x   = (const float*)d_in[0];
  const int*   ei  = (const int*)d_in[1];
  const int*   eli = (const int*)d_in[2];
  const float* Wc1 = (const float*)d_in[3];
  const float* bc1 = (const float*)d_in[4];
  const float* Wc2 = (const float*)d_in[5];
  const float* bc2 = (const float*)d_in[6];
  const float* W1  = (const float*)d_in[7];
  const float* b1  = (const float*)d_in[8];
  const float* W2  = (const float*)d_in[9];
  const float* b2  = (const float*)d_in[10];
  float* out = (float*)d_out;

  const int Nn = in_sizes[0] / 128;   // 50000
  const int Ee = in_sizes[1] / 2;     // 1.6M
  const int El = in_sizes[2] / 2;     // 200K

  // workspace layout (floats): dinv[N] | bufA[N*128] | bufB[N*128]
  //   bufA: xw1 -> xw2 in [0,N*64) -> z in-place ; agg2 in [N*64, N*128)
  //   bufB: agg1 -> h in-place
  float* dinv = (float*)d_ws;
  float* bufA = dinv + Nn;
  float* bufB = bufA + (size_t)Nn * 128;
  float* xw2  = bufA;
  float* agg2 = bufA + (size_t)Nn * 64;
  float* zz   = bufA;

  (void)hipMemsetAsync(dinv, 0, (size_t)Nn * sizeof(float), stream);
  (void)hipMemsetAsync(bufB, 0, (size_t)Nn * 128 * sizeof(float), stream);
  (void)hipMemsetAsync(agg2, 0, (size_t)Nn * 64 * sizeof(float), stream);

  const int* src = ei;
  const int* dst = ei + Ee;

  // degree + D^{-1/2}
  degree_kernel<<<ceil_div(Ee, 256), 256, 0, stream>>>(dst, dinv, Ee);
  finalize_dinv_kernel<<<ceil_div(Nn, 256), 256, 0, stream>>>(dinv, Nn);

  const int rowTiles = ceil_div(Nn, 16);
  const int gemmGx   = ceil_div(rowTiles, 4);

  // layer 1: xw1 = x @ Wc1 ; aggregate ; h = relu(agg + xw1*dinv^2 + bc1)
  gemm_wmma_kernel<128, 128><<<dim3(gemmGx, 8), 128, 0, stream>>>(x, Wc1, bufA, Nn);
  aggregate_kernel<128><<<8192, 256, 0, stream>>>(bufA, dinv, src, dst, bufB, Ee);
  combine_kernel<128, true><<<4096, 256, 0, stream>>>(bufB, bufA, dinv, bc1, bufB, Nn);

  // layer 2: xw2 = h @ Wc2 ; aggregate ; z = agg + xw2*dinv^2 + bc2
  gemm_wmma_kernel<128, 64><<<dim3(gemmGx, 4), 128, 0, stream>>>(bufB, Wc2, xw2, Nn);
  aggregate_kernel<64><<<8192, 256, 0, stream>>>(xw2, dinv, src, dst, agg2, Ee);
  combine_kernel<64, false><<<4096, 256, 0, stream>>>(agg2, xw2, dinv, bc2, zz, Nn);

  // fused decode MLP -> [2, EL]
  const int elTiles = ceil_div(El, 16);
  decode_kernel<<<ceil_div(elTiles, 4), 128, 0, stream>>>(zz, eli, W1, b1, W2, b2, out, El);
}
